// Net_26620207300758
// MI455X (gfx1250) — matrix operand order repaired and checked by reading, not compiled, not run on hardware
//
#include <hip/hip_runtime.h>
#include <hip/hip_bf16.h>

typedef __attribute__((ext_vector_type(16))) _Float16 v16h;
typedef __attribute__((ext_vector_type(8)))  _Float16 v8h;
typedef __attribute__((ext_vector_type(8)))  float    v8f;

#define NNODE 16384
#define DD    128
#define PKELEMS 16384   // packed f16 elements per 128x128 weight tile

// ---------------------------------------------------------------------------
// Pack a row-major f32 weight [Kreal x Nreal] into the CDNA5 WMMA B-fragment
// layout (padded to 128x128), so each lane loads 16 contiguous halves.
// index t = ((nt*4 + kt)*32 + lane)*16 + h
//   k = kt*32 + (lane&16 ? 8 : 0) + (h&7) + (h&8 ? 16 : 0) ;  n = nt*16 + (lane&15)
// ---------------------------------------------------------------------------
__global__ void k_pack_w(const float* __restrict__ W, _Float16* __restrict__ out,
                         int Kreal, int Nreal) {
  int t = blockIdx.x * blockDim.x + threadIdx.x;
  if (t >= PKELEMS) return;
  int h    = t & 15;
  int lane = (t >> 4) & 31;
  int kt   = (t >> 9) & 3;
  int nt   = t >> 11;
  int k = kt * 32 + ((lane & 16) ? 8 : 0) + (h & 7) + ((h & 8) ? 16 : 0);
  int n = nt * 16 + (lane & 15);
  float val = (k < Kreal && n < Nreal) ? W[k * Nreal + n] : 0.0f;
  out[t] = (_Float16)val;
}

// ---------------------------------------------------------------------------
// Generic WMMA GEMM: Y[M,128] = act(X[M,128] @ Wpk + bias)
// Block = 128 threads (4 waves), 64 rows per block, wave handles 16 rows.
// ---------------------------------------------------------------------------
__global__ void __launch_bounds__(128)
k_gemm(const float* __restrict__ X, const _Float16* __restrict__ Wpk,
       const float* __restrict__ bias, int nbias, float* __restrict__ Y,
       int relu) {
  __shared__ __align__(16) _Float16 tile[64][DD];
  const int tid   = threadIdx.x;
  const int lane  = tid & 31;
  const int wave  = tid >> 5;
  const int mbase = blockIdx.x * 64;

  // stage X tile -> f16 LDS (convert on the fly)
  for (int i = tid; i < 64 * DD; i += 128) {
    int r = i >> 7, c = i & 127;
    tile[r][c] = (_Float16)X[(size_t)(mbase + r) * DD + c];
  }
  __syncthreads();

  const int mrow = wave * 16;
  const _Float16* arow = &tile[mrow + (lane & 15)][0];

  // A fragments: 16x32 f16 layout (lanes 0-15: K {0-7,16-23}; 16-31: K {8-15,24-31})
  v16h a[4];
#pragma unroll
  for (int kt = 0; kt < 4; ++kt) {
    int off = kt * 32 + ((lane & 16) ? 8 : 0);
    v8h lo = *(const v8h*)(arow + off);
    v8h hi = *(const v8h*)(arow + off + 16);
#pragma unroll
    for (int q = 0; q < 8; ++q) { a[kt][q] = lo[q]; a[kt][8 + q] = hi[q]; }
  }

#pragma unroll
  for (int nt = 0; nt < 8; ++nt) {
    v8f acc = {};
#pragma unroll
    for (int kt = 0; kt < 4; ++kt) {
      v16h b = *(const v16h*)(Wpk + (size_t)((nt * 4 + kt) * 32 + lane) * 16);
      acc = __builtin_amdgcn_wmma_f32_16x16x32_f16(false, a[kt], false, b,
                                                   (short)0, acc, false, false);
    }
    int col = nt * 16 + (lane & 15);
    float bv = (bias != nullptr && col < nbias) ? bias[col] : 0.0f;
    int rbase = mbase + mrow + ((lane & 16) ? 8 : 0);
#pragma unroll
    for (int r = 0; r < 8; ++r) {
      float v = acc[r] + bv;
      if (relu) v = fmaxf(v, 0.0f);
      Y[(size_t)(rbase + r) * DD + col] = v;
    }
  }
}

// ---------------------------------------------------------------------------
// out[node] = dot(H[node][0..ncols-1], w) + b[0]   (wave per node)
// ---------------------------------------------------------------------------
__global__ void k_dot(const float* __restrict__ H, const float* __restrict__ w,
                      const float* __restrict__ b, float* __restrict__ out, int ncols) {
  int node = blockIdx.x * (blockDim.x >> 5) + (threadIdx.x >> 5);
  int lane = threadIdx.x & 31;
  if (node >= NNODE) return;
  const float* row = H + (size_t)node * DD;
  float s = 0.0f;
  for (int i = lane; i < ncols; i += 32) s += row[i] * w[i];
#pragma unroll
  for (int off = 16; off > 0; off >>= 1) s += __shfl_xor(s, off, 32);
  if (lane == 0) out[node] = s + b[0];
}

// ---------------------------------------------------------------------------
// Initial embedding: [rand(64) | mlp 2->61->61 | feat(2) | 0]
// ---------------------------------------------------------------------------
__global__ void k_embed(const float* __restrict__ feat, const float* __restrict__ rnd,
                        const float* __restrict__ w1, const float* __restrict__ b1,
                        const float* __restrict__ w2, const float* __restrict__ b2,
                        float* __restrict__ out) {
  int node = blockIdx.x * blockDim.x + threadIdx.x;
  if (node >= NNODE) return;
  float f0 = feat[node * 2 + 0], f1 = feat[node * 2 + 1];
  float h[61];
#pragma unroll 1
  for (int j = 0; j < 61; ++j) h[j] = fmaxf(f0 * w1[j] + f1 * w1[61 + j] + b1[j], 0.0f);
  float* o = out + (size_t)node * DD;
  for (int j = 0; j < 64; ++j) o[j] = rnd[(size_t)node * 64 + j];
#pragma unroll 1
  for (int j = 0; j < 61; ++j) {
    float s = b2[j];
    for (int k = 0; k < 61; ++k) s += h[k] * w2[k * 61 + j];
    o[64 + j] = s;
  }
  o[125] = f0; o[126] = f1; o[127] = 0.0f;
}

__global__ void k_zero_i(int* __restrict__ p, int n) {
  int i = blockIdx.x * blockDim.x + threadIdx.x;
  if (i < n) p[i] = 0;
}

__global__ void k_deg(const int* __restrict__ rows, int E, float* __restrict__ deg) {
  int e = blockIdx.x * blockDim.x + threadIdx.x;
  if (e < E) atomicAdd(deg + rows[e], 1.0f);
}

__global__ void k_zero_f(float* __restrict__ p, int n) {
  int i = blockIdx.x * blockDim.x + threadIdx.x;
  if (i < n) p[i] = 0.0f;
}

__global__ void k_inv(float* __restrict__ d, int n) {
  int i = blockIdx.x * blockDim.x + threadIdx.x;
  if (i < n) d[i] = 1.0f / d[i];
}

// ---------------------------------------------------------------------------
// CSR build (counting sort of edges by target column) — done ONCE, reused 6x.
// ---------------------------------------------------------------------------
__global__ void k_count(const int* __restrict__ cols, int E, int* __restrict__ counts) {
  int e = blockIdx.x * blockDim.x + threadIdx.x;
  if (e < E) atomicAdd(counts + cols[e], 1);
}

// Single block of 1024 threads scans 16384 counts -> exclusive offsets [NNODE+1]
__global__ void __launch_bounds__(1024)
k_scan(const int* __restrict__ counts, int* __restrict__ offs) {
  __shared__ int part[1024];
  int t = threadIdx.x;
  int base = t * 16;
  int local[16];
  int s = 0;
#pragma unroll
  for (int i = 0; i < 16; ++i) { local[i] = s; s += counts[base + i]; }
  part[t] = s;
  __syncthreads();
  for (int off = 1; off < 1024; off <<= 1) {
    int v = (t >= off) ? part[t - off] : 0;
    __syncthreads();
    part[t] += v;
    __syncthreads();
  }
  int pre = (t == 0) ? 0 : part[t - 1];
#pragma unroll
  for (int i = 0; i < 16; ++i) offs[base + i] = pre + local[i];
  if (t == 1023) offs[NNODE] = part[1023];
}

__global__ void k_copy_i(const int* __restrict__ src, int* __restrict__ dst, int n) {
  int i = blockIdx.x * blockDim.x + threadIdx.x;
  if (i < n) dst[i] = src[i];
}

__global__ void k_fill(const int* __restrict__ cols, int E,
                       int* __restrict__ cursor, int* __restrict__ eid) {
  int e = blockIdx.x * blockDim.x + threadIdx.x;
  if (e < E) {
    int c = cols[e];
    int p = atomicAdd(cursor + c, 1);
    eid[p] = e;
  }
}

// ---------------------------------------------------------------------------
// Fused v2c gather + combine (atomic-free): wave per target con node.
//   acc[j<127] = sum_e invdeg[r]*body[r][j] ; acc[127] = sum_e hv[r]*efv[r]
//   out = relu(acc + rc + bias | acc - rhs)
// ---------------------------------------------------------------------------
__global__ void k_gather_v2c(const int* __restrict__ eid, const int* __restrict__ offs,
                             const int* __restrict__ srcs,
                             const float* __restrict__ body, const float* __restrict__ hv,
                             const float* __restrict__ efv, const float* __restrict__ invdeg,
                             const float* __restrict__ rc, const float* __restrict__ bias,
                             const float* __restrict__ rhs, float* __restrict__ out) {
  int node = blockIdx.x * (blockDim.x >> 5) + (threadIdx.x >> 5);
  int lane = threadIdx.x & 31;
  if (node >= NNODE) return;
  float acc[4] = {0.f, 0.f, 0.f, 0.f};
  int s = offs[node], t = offs[node + 1];
  for (int k = s; k < t; ++k) {
    int e = eid[k];
    int r = srcs[e];
    if (k + 1 < t) __builtin_prefetch(body + (size_t)srcs[eid[k + 1]] * DD, 0, 1);
    float nrm = invdeg[r];
    const float* src = body + (size_t)r * DD;
#pragma unroll
    for (int q = 0; q < 4; ++q) {
      int j = lane + 32 * q;
      acc[q] += (j < 127) ? nrm * src[j] : hv[r] * efv[r];
    }
  }
#pragma unroll
  for (int q = 0; q < 4; ++q) {
    int j = lane + 32 * q;
    size_t idx = (size_t)node * DD + j;
    float v = (j < 127) ? (acc[q] + rc[idx] + bias[j]) : (acc[q] - rhs[node]);
    out[idx] = fmaxf(v, 0.0f);
  }
}

// ---------------------------------------------------------------------------
// Fused c2v gather + combine (atomic-free): wave per target var node.
//   viol_e = (efc[r]/asums[r]) * hv[node] * c[r][127]
//   acc[j<127] = sum_e invdeg[r]*body[r][j] ; acc[127] = sum_e invdeg[r]*viol_e
//   out = relu(acc + rc + bias | acc)
// ---------------------------------------------------------------------------
__global__ void k_gather_c2v(const int* __restrict__ eid, const int* __restrict__ offs,
                             const int* __restrict__ srcs,
                             const float* __restrict__ body, const float* __restrict__ hv,
                             const float* __restrict__ efc, const float* __restrict__ asums,
                             const float* __restrict__ cfeat, const float* __restrict__ invdeg,
                             const float* __restrict__ rc, const float* __restrict__ bias,
                             float* __restrict__ out) {
  int node = blockIdx.x * (blockDim.x >> 5) + (threadIdx.x >> 5);
  int lane = threadIdx.x & 31;
  if (node >= NNODE) return;
  float hvn = hv[node];
  float acc[4] = {0.f, 0.f, 0.f, 0.f};
  int s = offs[node], t = offs[node + 1];
  for (int k = s; k < t; ++k) {
    int e = eid[k];
    int r = srcs[e];
    if (k + 1 < t) __builtin_prefetch(body + (size_t)srcs[eid[k + 1]] * DD, 0, 1);
    float nrm  = invdeg[r];
    float viol = (efc[r] / asums[r]) * hvn * cfeat[(size_t)r * DD + 127];
    const float* src = body + (size_t)r * DD;
#pragma unroll
    for (int q = 0; q < 4; ++q) {
      int j = lane + 32 * q;
      acc[q] += nrm * ((j < 127) ? src[j] : viol);
    }
  }
#pragma unroll
  for (int q = 0; q < 4; ++q) {
    int j = lane + 32 * q;
    size_t idx = (size_t)node * DD + j;
    float v = (j < 127) ? (acc[q] + rc[idx] + bias[j]) : acc[q];
    out[idx] = fmaxf(v, 0.0f);
  }
}

// ===========================================================================
// Host side
// ===========================================================================
extern "C" void kernel_launch(void* const* d_in, const int* in_sizes, int n_in,
                              void* d_out, int out_size, void* d_ws, size_t ws_size,
                              hipStream_t stream) {
  // ---- data inputs ----
  const float* var_feat = (const float*)d_in[0];
  const float* con_feat = (const float*)d_in[1];
  const float* ef_var   = (const float*)d_in[2];
  const float* ef_con   = (const float*)d_in[3];
  const float* rhs      = (const float*)d_in[4];
  const float* asums    = (const float*)d_in[5];
  const float* rand_var = (const float*)d_in[6];
  const float* rand_con = (const float*)d_in[7];
  const int*   ei_var   = (const int*)d_in[8];   // [0..E): src var ; [E..2E): dst con
  const int*   ei_con   = (const int*)d_in[9];   // [0..E): src con ; [E..2E): dst var
  const int    E        = in_sizes[8] / 2;

  // ---- param leaves (sorted-dict tree flatten order) ----
  auto F = [&](int i) { return (const float*)d_in[i]; };
  // 10..13 con_mlp {l1.b,l1.w,l2.b,l2.w} ; 14..25 fc1..fc6 {b,w}
  const int PL = 26;
  enum { C2V_BIAS = 0, C2V_L1B, C2V_L1W, C2V_L2B, C2V_L2W, C2V_ROOT,
         H2V_L1B, H2V_L1W, H2V_L2B, H2V_L2W,
         V2C_BIAS, V2C_L1B, V2C_L1W, V2C_L2B, V2C_L2W, V2C_ROOT };
  // 122..125: var_mlp {l1.b,l1.w,l2.b,l2.w}

  // ---- workspace carve ----
  const size_t MAT = (size_t)NNODE * DD;           // 2,097,152 floats
  float* wsf   = (float*)d_ws;
  float* Vb    = wsf;                 // node features (vars)
  float* Cb    = wsf + MAT;           // node features (cons)
  float* bufA  = wsf + 2 * MAT;       // hidden / ping
  float* bufB  = wsf + 3 * MAT;       // mlp body / pong
  float* rc    = wsf + 4 * MAT;       // @root term
  float* hv    = wsf + 5 * MAT;                     // NNODE
  float* idv   = hv  + NNODE;                       // inv deg (var side)
  float* idc   = idv + NNODE;                       // inv deg (con side)
  _Float16* pk = (_Float16*)(idc + NNODE);          // packed weights: 48 slots
  auto slot = [&](int s) { return pk + (size_t)s * PKELEMS; };
  int* ints   = (int*)(pk + (size_t)48 * PKELEMS);
  int* offs_v = ints;                 // [NNODE+1]  v2c CSR offsets (targets = cons)
  int* offs_c = offs_v + NNODE + 1;   // [NNODE+1]  c2v CSR offsets (targets = vars)
  int* cur    = offs_c + NNODE + 1;   // [NNODE]    cursor / counts (reused)
  int* eid_v  = cur + NNODE;          // [E]        edge ids sorted by con target
  int* eid_c  = eid_v + E;            // [E]        edge ids sorted by var target

  const dim3 B256(256);
  const int GB_N   = (NNODE + 255) / 256;
  const int GB_GEM = NNODE / 64;
  const int GB_W   = NNODE / 8;                    // wave-per-node (8 waves/blk)
  const int GB_EL  = (E + 255) / 256;

  // ---- pack all weights to f16 WMMA-B layout (once) ----
  for (int i = 0; i < 6; ++i) {
    int L = PL + 16 * i;
    k_pack_w<<<64, B256, 0, stream>>>(F(L + V2C_L1W), slot(7 * i + 0), 128, 127);
    k_pack_w<<<64, B256, 0, stream>>>(F(L + V2C_L2W), slot(7 * i + 1), 127, 127);
    k_pack_w<<<64, B256, 0, stream>>>(F(L + V2C_ROOT), slot(7 * i + 2), 128, 127);
    k_pack_w<<<64, B256, 0, stream>>>(F(L + C2V_L1W), slot(7 * i + 3), 128, 127);
    k_pack_w<<<64, B256, 0, stream>>>(F(L + C2V_L2W), slot(7 * i + 4), 127, 127);
    k_pack_w<<<64, B256, 0, stream>>>(F(L + C2V_ROOT), slot(7 * i + 5), 128, 127);
    k_pack_w<<<64, B256, 0, stream>>>(F(L + H2V_L1W), slot(7 * i + 6), 128, 127);
  }
  for (int k = 0; k < 5; ++k)  // fc1..fc5 (w at 15,17,19,21,23)
    k_pack_w<<<64, B256, 0, stream>>>(F(15 + 2 * k), slot(42 + k), 128, 128);

  // ---- degree normalization (source-degree, per direction) ----
  k_zero_f<<<GB_N, B256, 0, stream>>>(idv, NNODE);
  k_zero_f<<<GB_N, B256, 0, stream>>>(idc, NNODE);
  k_deg<<<GB_EL, B256, 0, stream>>>(ei_var, E, idv);
  k_deg<<<GB_EL, B256, 0, stream>>>(ei_con, E, idc);
  k_inv<<<GB_N, B256, 0, stream>>>(idv, NNODE);
  k_inv<<<GB_N, B256, 0, stream>>>(idc, NNODE);

  // ---- build CSR by target, once per direction (reused by all 6 layers) ----
  // v2c: targets are cons = ei_var[E..2E)
  k_zero_i<<<GB_N, B256, 0, stream>>>(cur, NNODE);
  k_count<<<GB_EL, B256, 0, stream>>>(ei_var + E, E, cur);
  k_scan<<<1, 1024, 0, stream>>>(cur, offs_v);
  k_copy_i<<<GB_N, B256, 0, stream>>>(offs_v, cur, NNODE);
  k_fill<<<GB_EL, B256, 0, stream>>>(ei_var + E, E, cur, eid_v);
  // c2v: targets are vars = ei_con[E..2E)
  k_zero_i<<<GB_N, B256, 0, stream>>>(cur, NNODE);
  k_count<<<GB_EL, B256, 0, stream>>>(ei_con + E, E, cur);
  k_scan<<<1, 1024, 0, stream>>>(cur, offs_c);
  k_copy_i<<<GB_N, B256, 0, stream>>>(offs_c, cur, NNODE);
  k_fill<<<GB_EL, B256, 0, stream>>>(ei_con + E, E, cur, eid_c);

  // ---- initial embeddings ----
  k_embed<<<GB_N, B256, 0, stream>>>(var_feat, rand_var, F(123), F(122), F(125), F(124), Vb);
  k_embed<<<GB_N, B256, 0, stream>>>(con_feat, rand_con, F(11),  F(10),  F(13),  F(12),  Cb);

  // ---- 6 message-passing layers ----
  for (int i = 0; i < 6; ++i) {
    int L = PL + 16 * i;
    // hv = mlp(h2v, v)[:,0]  (node-level; serves both v2c and c2v of this layer)
    k_gemm<<<GB_GEM, 128, 0, stream>>>(Vb, slot(7 * i + 6), F(L + H2V_L1B), 127, bufA, 1);
    k_dot<<<GB_W, B256, 0, stream>>>(bufA, F(L + H2V_L2W), F(L + H2V_L2B), hv, 127);

    // ---- v2c ----
    k_gemm<<<GB_GEM, 128, 0, stream>>>(Vb, slot(7 * i + 0), F(L + V2C_L1B), 127, bufA, 1);
    k_gemm<<<GB_GEM, 128, 0, stream>>>(bufA, slot(7 * i + 1), F(L + V2C_L2B), 127, bufB, 0);
    k_gemm<<<GB_GEM, 128, 0, stream>>>(Cb, slot(7 * i + 2), nullptr, 0, rc, 0);  // c_old @ root
    k_gather_v2c<<<GB_W, B256, 0, stream>>>(eid_v, offs_v, ei_var, bufB, hv,
                                            ef_var, idv, rc, F(L + V2C_BIAS), rhs, Cb);

    // ---- c2v ----
    k_gemm<<<GB_GEM, 128, 0, stream>>>(Cb, slot(7 * i + 3), F(L + C2V_L1B), 127, bufA, 1);
    k_gemm<<<GB_GEM, 128, 0, stream>>>(bufA, slot(7 * i + 4), F(L + C2V_L2B), 127, bufB, 0);
    k_gemm<<<GB_GEM, 128, 0, stream>>>(Vb, slot(7 * i + 5), nullptr, 0, rc, 0);  // v_old @ root
    k_gather_c2v<<<GB_W, B256, 0, stream>>>(eid_c, offs_c, ei_con, bufB, hv,
                                            ef_con, asums, Cb, idc, rc, F(L + C2V_BIAS), Vb);
  }

  // ---- FC head: 5x (gemm + relu) then 128-dot ----
  k_gemm<<<GB_GEM, 128, 0, stream>>>(Vb,   slot(42), F(14), 128, bufA, 1);
  k_gemm<<<GB_GEM, 128, 0, stream>>>(bufA, slot(43), F(16), 128, bufB, 1);
  k_gemm<<<GB_GEM, 128, 0, stream>>>(bufB, slot(44), F(18), 128, bufA, 1);
  k_gemm<<<GB_GEM, 128, 0, stream>>>(bufA, slot(45), F(20), 128, bufB, 1);
  k_gemm<<<GB_GEM, 128, 0, stream>>>(bufB, slot(46), F(22), 128, bufA, 1);
  k_dot<<<GB_W, B256, 0, stream>>>(bufA, F(25), F(24), (float*)d_out, 128);
}
